// SGP_LSTM_84636625535586
// MI455X (gfx1250) — matrix operand order repaired and checked by reading, not compiled, not run on hardware
//
#include <hip/hip_runtime.h>
#include <hip/hip_bf16.h>
#include <math.h>

typedef __attribute__((ext_vector_type(16))) __bf16 v16bf;
typedef __attribute__((ext_vector_type(8)))  float  v8f;

#define BB 64
#define SS 1024
#define DD 502
#define HH 512
#define PP 10
#define DP 512            // D + P
#define G4 2048           // 4*H
#define M_ROWS (BB * SS)  // 65536

// round-to-nearest-even fp32 -> bf16
__device__ __forceinline__ unsigned short f2bf(float f) {
    union { float f; unsigned u; } v; v.f = f;
    unsigned r = v.u + 0x7FFFu + ((v.u >> 16) & 1u);
    return (unsigned short)(r >> 16);
}

__device__ __forceinline__ float sigf(float x) {
    return 1.0f / (1.0f + __expf(-x));
}

// ---------------------------------------------------------------------------
// Kernel 1: symbolic-program features + pack combined[(s*B+b), 0..511] as bf16.
// ---------------------------------------------------------------------------
__global__ void __launch_bounds__(128)
prep_kernel(const float* __restrict__ x, const float* __restrict__ pw,
            unsigned short* __restrict__ comb) {
    const int idx = blockIdx.x;          // b*S + s (x layout)
    const int b = idx / SS, s = idx % SS;
    const int m = s * BB + b;            // time-major row
    const float* xr = x + (size_t)idx * DD;
    unsigned short* cr = comb + (size_t)m * DP;

    for (int d = threadIdx.x; d < DD; d += blockDim.x)
        cr[d] = f2bf(xr[d]);

    const int t = threadIdx.x;
    if (t < PP) {
        float v = 0.0f;
        switch (t) {
            case 0: v = __sinf(xr[0] + xr[1]); break;
            case 1: v = xr[2] * xr[3] + 0.5f; break;
            case 2: v = __logf(fabsf(xr[4]) + 1e-8f); break;
            case 3: v = __sinf(xr[5] * 0.7f); break;
            case 4: v = (xr[6] + xr[7]) * xr[8]; break;
            case 5: v = __logf(fabsf(xr[9] + xr[10]) + 1e-8f); break;
            case 6: v = __sinf(xr[11]) + xr[12]; break;
            case 7: v = xr[13] * __sinf(xr[14]); break;
            case 8: v = __logf(fabsf(xr[15] * xr[16]) + 1e-8f); break;
            case 9: v = xr[17] - 0.25f; break;
        }
        cr[DD + t] = f2bf(v * pw[t]);
    }
}

// ---------------------------------------------------------------------------
// Kernel 2: pack W_ih / W_hh to bf16, bias = b_ih + b_hh
// ---------------------------------------------------------------------------
__global__ void __launch_bounds__(256)
pack_weights_kernel(const float* __restrict__ Wih, const float* __restrict__ Whh,
                    const float* __restrict__ bih, const float* __restrict__ bhh,
                    unsigned short* __restrict__ Wihbf,
                    unsigned short* __restrict__ Whhbf,
                    float* __restrict__ bias) {
    const int i = blockIdx.x * blockDim.x + threadIdx.x;
    if (i < G4 * HH) {
        Wihbf[i] = f2bf(Wih[i]);
        Whhbf[i] = f2bf(Whh[i]);
    }
    if (i < G4) bias[i] = bih[i] + bhh[i];
}

// ---------------------------------------------------------------------------
// Kernel 3: xW[m, n] = combined[m, :]·W_ih[n, :] + bias[n]
// M=65536, N=2048, K=512.  Per wave: one 16(M) x 64(N) strip.
// Double-buffered A/B; sched_barrier(0) pins the prefetch block ahead of the
// WMMA block so the scheduler cannot sink loads onto their consumers.
// ---------------------------------------------------------------------------
__global__ void __launch_bounds__(256)
gemm_xw_kernel(const unsigned short* __restrict__ Abf,
               const unsigned short* __restrict__ Wihbf,
               const float* __restrict__ bias,
               float* __restrict__ xW) {
    const int wid  = blockIdx.x * 8 + (threadIdx.x >> 5);
    const int lane = threadIdx.x & 31;
    const int lr = lane & 15, hi = lane >> 4;
    const int tn = wid & 31;         // 32 N-tiles of 64
    const int tm = wid >> 5;         // 4096 M-tiles of 16
    const int m0 = tm * 16, n0 = tn * 64;

    v8f acc[4];
    #pragma unroll
    for (int j = 0; j < 4; ++j)
        #pragma unroll
        for (int r = 0; r < 8; ++r) acc[j][r] = 0.0f;

    // Per-lane base pointers (A 16x32 fragment, B 32x16 fragment layouts).
    const unsigned short* arow = Abf + (size_t)(m0 + lr) * DP + hi * 16;
    const unsigned short* brow = Wihbf + (size_t)(n0 + lr) * DP + hi * 16;

    union AF { v16bf v; uint4 q[2]; };
    AF    a[2];
    v16bf b[2][4];

    // prologue: fragments for k0 = 0
    a[0].q[0] = *(const uint4*)(arow + 0);
    a[0].q[1] = *(const uint4*)(arow + 8);
    #pragma unroll
    for (int j = 0; j < 4; ++j)
        b[0][j] = *(const v16bf*)(brow + (size_t)j * 16 * DP);

    #pragma unroll 2
    for (int k0 = 0; k0 < DP; k0 += 32) {
        const int cur = (k0 >> 5) & 1;
        const int nxt = cur ^ 1;
        // unconditional next-k prefetch (tail lands in workspace guard region)
        a[nxt].q[0] = *(const uint4*)(arow + k0 + 32);
        a[nxt].q[1] = *(const uint4*)(arow + k0 + 40);
        #pragma unroll
        for (int j = 0; j < 4; ++j)
            b[nxt][j] = *(const v16bf*)(brow + (size_t)j * 16 * DP + k0 + 32);
        __builtin_amdgcn_sched_barrier(0);   // loads may not sink past here
        #pragma unroll
        for (int j = 0; j < 4; ++j)
            acc[j] = __builtin_amdgcn_wmma_f32_16x16x32_bf16(
                false, a[cur].v, false, b[cur][j], (short)0, acc[j], false, false);
        __builtin_amdgcn_sched_barrier(0);   // WMMAs may not sink below next loads
    }

    #pragma unroll
    for (int j = 0; j < 4; ++j) {
        const int col = n0 + j * 16 + lr;
        const float bb = bias[col];
        #pragma unroll
        for (int r = 0; r < 8; ++r) {
            const int row = m0 + r + hi * 8;   // C layout: M = r + 8*hi
            xW[(size_t)row * G4 + col] = acc[j][r] + bb;
        }
    }
}

// ---------------------------------------------------------------------------
// Kernel 4: LSTM scan.  4 blocks (16 batch rows each) — batch rows are
// independent, so the 1024-step scan needs only workgroup barriers.
// 16 waves per block, each owning a 16x128 gate strip (8 acc tiles).
// B = W_hh(bf16) fragment stream (frag f = 8k + j) with quad-buffer,
// prefetch depth 3, sched_barrier(0) per WMMA: every WMMA runs with 3
// fragment loads in flight.  A = h(bf16) from LDS via ds_read_b128.
// ---------------------------------------------------------------------------
__global__ void __launch_bounds__(512)
lstm_kernel(const float* __restrict__ xW,
            const unsigned short* __restrict__ Whhbf,
            const float* __restrict__ Wout, const float* __restrict__ bout,
            float* __restrict__ out) {
    extern __shared__ char smem[];
    float* Gs = (float*)smem;                                   // [16][2048] raw gates
    float* cs = (float*)(smem + 16 * G4 * 4);                   // [16][512] cell fp32
    float* hs = (float*)(smem + 16 * G4 * 4 + 16 * HH * 4);     // [16][512] hidden fp32
    unsigned short* hbf =
        (unsigned short*)(smem + 16 * G4 * 4 + 2 * 16 * HH * 4); // [16][512] hidden bf16

    const int bg   = blockIdx.x;        // batch group: rows bg*16 .. bg*16+15
    const int tid  = threadIdx.x;
    const int wave = tid >> 5, lane = tid & 31;
    const int lr = lane & 15, hi = lane >> 4;
    const int n0 = wave * 128;          // this wave's gate-column strip

    for (int i = tid; i < 16 * HH; i += 512) {
        cs[i] = 0.0f; hs[i] = 0.0f; hbf[i] = 0;
    }
    __syncthreads();

    // Per-lane base pointers for the wave's W_hh strip / h rows.
    const unsigned short* bbase = Whhbf + (size_t)(n0 + lr) * HH + hi * 16;
    const unsigned short* ar    = hbf + lr * HH + hi * 16;

    for (int t = 0; t < SS; ++t) {
        v8f acc[8];
        #pragma unroll
        for (int j = 0; j < 8; ++j)
            #pragma unroll
            for (int r = 0; r < 8; ++r) acc[j][r] = 0.0f;

        union AF { v16bf v; uint4 q[2]; };
        v16bf b[4];
        // prologue: fragments f = 0,1,2  (frag f: column tile f&7, k = (f>>3)*32)
        b[0] = *(const v16bf*)(bbase);
        b[1] = *(const v16bf*)(bbase + (size_t)1 * 16 * HH);
        b[2] = *(const v16bf*)(bbase + (size_t)2 * 16 * HH);

        for (int k0 = 0; k0 < HH; k0 += 32) {
            AF a;
            a.q[0] = *(const uint4*)(ar + k0);
            a.q[1] = *(const uint4*)(ar + k0 + 8);
            #pragma unroll
            for (int jj = 0; jj < 8; ++jj) {
                // prefetch frag f+3 = 8k + jj + 3 into buffer (jj+3)&3
                const int pj = (jj + 3) & 7;                 // column tile of f+3
                const int pk = (jj >= 5) ? 32 : 0;           // k advance of f+3
                b[(jj + 3) & 3] =
                    *(const v16bf*)(bbase + (size_t)pj * 16 * HH + k0 + pk);
                __builtin_amdgcn_sched_barrier(0);           // keep 3 loads in flight
                acc[jj] = __builtin_amdgcn_wmma_f32_16x16x32_bf16(
                    false, a.v, false, b[jj & 3], (short)0, acc[jj], false, false);
                __builtin_amdgcn_sched_barrier(0);
            }
        }

        // G = h@W_hh^T + xW_t   -> LDS
        const float* xwt = xW + ((size_t)t * BB + bg * 16) * G4;
        #pragma unroll
        for (int j = 0; j < 8; ++j) {
            const int col = n0 + j * 16 + lr;
            #pragma unroll
            for (int r = 0; r < 8; ++r) {
                const int row = r + hi * 8;
                Gs[row * G4 + col] = acc[j][r] + xwt[(size_t)row * G4 + col];
            }
        }
        __syncthreads();

        // gate activations + state update (16 elements per thread)
        for (int e = tid; e < 16 * HH; e += 512) {
            const int mrow = e >> 9;       // /512
            const int jcol = e & 511;
            const float gi = Gs[mrow * G4 + jcol];
            const float gf = Gs[mrow * G4 + jcol + 512];
            const float gg = Gs[mrow * G4 + jcol + 1024];
            const float go = Gs[mrow * G4 + jcol + 1536];
            const float ii = sigf(gi);
            const float ff = sigf(gf);
            const float g  = tanhf(gg);
            const float oo = sigf(go);
            const float c  = ff * cs[e] + ii * g;
            cs[e] = c;
            const float h = oo * tanhf(c);
            hs[e] = h;
            hbf[e] = f2bf(h);
        }
        __syncthreads();
    }

    // out[b] = h_last[b,:]·W_out + b_out
    if (tid < 16) {
        float accd = bout[0];
        const float* hr = hs + tid * HH;
        for (int j = 0; j < HH; ++j) accd += hr[j] * Wout[j];
        out[bg * 16 + tid] = accd;
    }
}

// ---------------------------------------------------------------------------
extern "C" void kernel_launch(void* const* d_in, const int* in_sizes, int n_in,
                              void* d_out, int out_size, void* d_ws, size_t ws_size,
                              hipStream_t stream) {
    const float* x    = (const float*)d_in[0];
    const float* pw   = (const float*)d_in[1];
    const float* Wih  = (const float*)d_in[2];
    const float* Whh  = (const float*)d_in[3];
    const float* bih  = (const float*)d_in[4];
    const float* bhh  = (const float*)d_in[5];
    const float* Wout = (const float*)d_in[6];
    const float* bout = (const float*)d_in[7];
    float* out = (float*)d_out;

    char* ws = (char*)d_ws;
    size_t off = 0;
    auto take = [&](size_t bytes) -> char* {
        char* p = ws + off;
        off = (off + bytes + 255) & ~(size_t)255;
        return p;
    };
    unsigned short* comb  = (unsigned short*)take((size_t)M_ROWS * DP * 2);  // 64 MB
    unsigned short* Wihbf = (unsigned short*)take((size_t)G4 * HH * 2);      // 2 MB
    unsigned short* Whhbf = (unsigned short*)take((size_t)G4 * HH * 2);      // 2 MB
    float*          bias  = (float*)take((size_t)G4 * 4);                    // 8 KB
    float*          xW    = (float*)take((size_t)M_ROWS * G4 * 4);           // 512 MB
    (void)take(65536);  // guard region for tail prefetches

    pack_weights_kernel<<<(G4 * HH + 255) / 256, 256, 0, stream>>>(
        Wih, Whh, bih, bhh, Wihbf, Whhbf, bias);

    prep_kernel<<<M_ROWS, 128, 0, stream>>>(x, pw, comb);

    // 4096 M-tiles x 32 N-tiles = 131072 wave-tiles, 8 waves per block
    gemm_xw_kernel<<<131072 / 8, 256, 0, stream>>>(comb, Wihbf, bias, xW);

    const size_t lstm_lds = (size_t)16 * G4 * 4 + 2 * (size_t)16 * HH * 4
                          + (size_t)16 * HH * 2;   // 208 KB of the 320 KB WGP LDS
    lstm_kernel<<<BB / 16, 512, lstm_lds, stream>>>(xW, Whhbf, Wout, bout, out);
}
